// VQ_24893630448037
// MI455X (gfx1250) — matrix operand order repaired and checked by reading, not compiled, not run on hardware
//
#include <hip/hip_runtime.h>

// ---- CDNA5 (gfx1250, wave32) VQ kernel set -------------------------------
// reference: x_in [64,256,1024] f32, emb [1024,256] f32
// outputs (concat in d_out, float): x_q_out [64,256,1024], idxs [64,1024], vq_loss [1]

typedef __attribute__((ext_vector_type(16))) _Float16 v16h;
typedef __attribute__((ext_vector_type(8)))  _Float16 v8h;
typedef __attribute__((ext_vector_type(4)))  _Float16 v4h;
typedef __attribute__((ext_vector_type(8)))  float    v8f;
typedef __attribute__((ext_vector_type(4)))  unsigned int u32x4;
typedef __attribute__((ext_vector_type(8)))  int      i32x8;
typedef __attribute__((ext_vector_type(4)))  int      i32x4;

constexpr int Bn = 64, Cn = 256, Ln = 1024, En = 1024;
constexpr int Nn = Bn * Ln;                 // 65536 rows
constexpr int ROWS_PER_BLOCK = 128;         // 8 waves * 16 rows
constexpr int APAD = 264;                   // 256 + 8 halfs (16B) pad -> conflict-free LDS
constexpr int GATHER_BLOCKS = Nn / 64;      // 1024
constexpr int TILE_HALFS = 16 * Cn;         // 4096 halves per code tile per (hi|lo)

#if defined(__has_builtin)
#if __has_builtin(__builtin_amdgcn_tensor_load_to_lds) && \
    __has_builtin(__builtin_amdgcn_s_wait_tensorcnt)
#define USE_TDM 1
#endif
#endif
#ifndef USE_TDM
#define USE_TDM 0
#endif

// ---------------------------------------------------------------------------
// Kernel 0a: per-code squared norms
__global__ __launch_bounds__(256) void vq_norms(const float* __restrict__ emb,
                                                float* __restrict__ enorm) {
  int j = blockIdx.x * 256 + threadIdx.x;          // 0..1023
  const float4* row = (const float4*)(emb + (size_t)j * Cn);
  float s = 0.f;
#pragma unroll 8
  for (int i = 0; i < Cn / 4; ++i) {
    float4 v = row[i];
    s += v.x * v.x + v.y * v.y + v.z * v.z + v.w * v.w;
  }
  enorm[j] = s;
}

// Kernel 0b: one-time split of the codebook into f16 hi/lo planes (L2-resident)
__global__ __launch_bounds__(256) void vq_split(const float* __restrict__ emb,
                                                _Float16* __restrict__ eh,
                                                _Float16* __restrict__ el) {
  int g = blockIdx.x * 256 + threadIdx.x;          // 65536 float4 groups
  float4 v = ((const float4*)emb)[g];
  v4h hh, ll;
  hh.x = (_Float16)v.x; hh.y = (_Float16)v.y;
  hh.z = (_Float16)v.z; hh.w = (_Float16)v.w;
  ll.x = (_Float16)(v.x - (float)hh.x); ll.y = (_Float16)(v.y - (float)hh.y);
  ll.z = (_Float16)(v.z - (float)hh.z); ll.w = (_Float16)(v.w - (float)hh.w);
  *(v4h*)(eh + (size_t)g * 4) = hh;
  *(v4h*)(el + (size_t)g * 4) = ll;
}

// ---------------------------------------------------------------------------
#if USE_TDM
// Issue one TDM 2-D tile load: 16 rows x 512B (f16) global -> LDS, with a
// 16B pad inserted after every 512B row (=> LDS row stride APAD halves).
// D# layout per CDNA5 ISA 8.3/8.4.
__device__ __forceinline__ void tdm_issue_tile(const _Float16* gsrc, unsigned lds_off) {
  unsigned long long ga = (unsigned long long)(uintptr_t)gsrc;
  u32x4 g0;
  g0[0] = 1u;                                            // count=1 valid descriptor
  g0[1] = lds_off;                                       // lds_addr
  g0[2] = (unsigned)ga;                                  // global_addr[31:0]
  g0[3] = (unsigned)((ga >> 32) & 0x01FFFFFFu) | (2u << 30);  // addr[56:32] | type=2
  i32x8 g1;
  g1[0] = (int)((1u << 16)      // data_size = 2 bytes
              | (1u << 20)      // pad_enable
              | (6u << 22)      // pad_interval: 128 DWORDs (512B) between pads
              | (3u << 25));    // pad_amount: 4 DWORDs (16B)
  g1[1] = (int)(256u << 16);    // tensor_dim0[15:0] = 256 (bits 63:48)
  g1[2] = (int)(16u << 16);     // tensor_dim1[15:0] = 16  (bits 95:80)
  g1[3] = (int)(256u << 16);    // tile_dim0 = 256         (bits 127:112)
  g1[4] = 16;                   // tile_dim1 = 16, tile_dim2 = 0 (2-D)
  g1[5] = 256;                  // tensor_dim0_stride = 256 elements
  g1[6] = 0;
  g1[7] = 0;
  i32x4 g2 = {0, 0, 0, 0};
  i32x4 g3 = {0, 0, 0, 0};
#if __clang_major__ >= 23
  i32x8 g4 = {0, 0, 0, 0, 0, 0, 0, 0};
  __builtin_amdgcn_tensor_load_to_lds(g0, g1, g2, g3, g4, 0);
#else
  __builtin_amdgcn_tensor_load_to_lds(g0, g1, g2, g3, 0);
#endif
}
#endif

// Synchronous staging fallback (also used when TDM builtin is unavailable)
__device__ __forceinline__ void stage_tile_sync(const _Float16* __restrict__ gh,
                                                const _Float16* __restrict__ gl,
                                                _Float16* sh, _Float16* sl, int t) {
#pragma unroll
  for (int i = 0; i < 2; ++i) {
    int slot = t + i * 256;                // 512 chunks of 8 halves per plane
    int row = slot >> 5, c8 = slot & 31;
    v8h vh = *(const v8h*)(gh + (size_t)slot * 8);
    v8h vl = *(const v8h*)(gl + (size_t)slot * 8);
    *(v8h*)&sh[row * APAD + c8 * 8] = vh;
    *(v8h*)&sl[row * APAD + c8 * 8] = vl;
  }
}

// ---------------------------------------------------------------------------
// Kernel 1: distance GEMM (split-f16 WMMA, f32 accumulate) + running argmin.
// A (SRC0) = 16-code tile (M=codes); B (SRC1) = 16 x-rows (N on contiguous l).
// TDM double-buffers the next code tile under the WMMA stream; A fragments are
// software-pipelined through two register buffers so DS latency hides under
// the XDL pipe instead of forcing full s_wait_dscnt drains.
__global__ __launch_bounds__(256) void vq_argmin(const float* __restrict__ x,
                                                 const _Float16* __restrict__ eh,
                                                 const _Float16* __restrict__ el,
                                                 const float* __restrict__ enorm,
                                                 int* __restrict__ idx_ws,
                                                 float* __restrict__ idx_out) {
  __shared__ _Float16 lds_ah[2][16 * APAD];
  __shared__ _Float16 lds_al[2][16 * APAD];
  __shared__ float    lds_norm[En];

  const int t    = threadIdx.x;
  const int wid  = t >> 5;
  const int lane = t & 31;
  const int nlo  = lane & 15;
  const int hi   = lane >> 4;

#if USE_TDM
  const unsigned ah_off[2] = {(unsigned)(uintptr_t)&lds_ah[0][0],
                              (unsigned)(uintptr_t)&lds_ah[1][0]};
  const unsigned al_off[2] = {(unsigned)(uintptr_t)&lds_al[0][0],
                              (unsigned)(uintptr_t)&lds_al[1][0]};
  if (wid == 0) {                           // prologue: DMA tile 0 into buffer 0
    tdm_issue_tile(eh, ah_off[0]);
    tdm_issue_tile(el, al_off[0]);
  }
#else
  stage_tile_sync(eh, el, &lds_ah[0][0], &lds_al[0][0], t);
#endif

  // stage all code norms once (4 KB) — overlaps with tile-0 DMA
#pragma unroll
  for (int i = 0; i < 4; ++i) lds_norm[t + i * 256] = enorm[t + i * 256];

  // ---- register-cache B fragments (this wave's 16 x-rows, full K=256) ----
  // B 32x16 f16 layout: component j -> K = 8*hi + (j&7) + 16*(j>>3), N = lane%16
  const int n0 = blockIdx.x * ROWS_PER_BLOCK + wid * 16;  // one b per block
  const int b  = n0 >> 10;
  const int l  = (n0 & (Ln - 1)) + nlo;
  const float* xb = x + (size_t)b * Cn * Ln + l;          // xb[k*Ln] = x[b,k,l]

  v16h bh[8], bl[8];
#pragma unroll
  for (int c = 0; c < 8; ++c) {
#pragma unroll
    for (int j = 0; j < 16; ++j) {
      float xv = xb[(size_t)(32 * c + 8 * hi + (j & 7) + ((j >> 3) << 4)) * Ln];
      _Float16 h = (_Float16)xv;
      bh[c][j] = h;
      bl[c][j] = (_Float16)(xv - (float)h);
    }
  }

#if USE_TDM
  if (wid == 0) __builtin_amdgcn_s_wait_tensorcnt(0);
#endif
  __syncthreads();

  float best_d = 3.4e38f;
  int   best_i = 0;

  for (int mt = 0; mt < En / 16; ++mt) {
    const int cur = mt & 1;
    // kick DMA / staging for tile mt+1 into the other buffer (WAR-safe: that
    // buffer was last read in iteration mt-1, before the previous barrier)
    if (mt + 1 < En / 16) {
#if USE_TDM
      if (wid == 0) {
        tdm_issue_tile(eh + (size_t)(mt + 1) * TILE_HALFS, ah_off[cur ^ 1]);
        tdm_issue_tile(el + (size_t)(mt + 1) * TILE_HALFS, al_off[cur ^ 1]);
      }
#else
      stage_tile_sync(eh + (size_t)(mt + 1) * TILE_HALFS,
                      el + (size_t)(mt + 1) * TILE_HALFS,
                      &lds_ah[cur ^ 1][0], &lds_al[cur ^ 1][0], t);
#endif
    }

    // A 16x32 f16 layout: lane holds M=nlo; per chunk c the two 8-half runs
    // sit at  nlo*APAD + 32*c + 8*hi  and +16.  Pipeline chunk c+1's loads
    // under chunk c's 3 WMMAs via two register buffers.
    const _Float16* pah = &lds_ah[cur][nlo * APAD + 8 * hi];
    const _Float16* pal = &lds_al[cur][nlo * APAD + 8 * hi];

    v8h ah0[2], ah1[2], al0[2], al1[2];
    ah0[0] = *(const v8h*)(pah + 0);
    ah1[0] = *(const v8h*)(pah + 16);
    al0[0] = *(const v8h*)(pal + 0);
    al1[0] = *(const v8h*)(pal + 16);

    v8f acc = {};
#pragma unroll
    for (int c = 0; c < 8; ++c) {
      const int cb = c & 1, nb = cb ^ 1;
      if (c + 1 < 8) {
        ah0[nb] = *(const v8h*)(pah + 32 * (c + 1));
        ah1[nb] = *(const v8h*)(pah + 32 * (c + 1) + 16);
        al0[nb] = *(const v8h*)(pal + 32 * (c + 1));
        al1[nb] = *(const v8h*)(pal + 32 * (c + 1) + 16);
      }
      v16h ah = __builtin_shufflevector(ah0[cb], ah1[cb],
                                        0,1,2,3,4,5,6,7,8,9,10,11,12,13,14,15);
      v16h al = __builtin_shufflevector(al0[cb], al1[cb],
                                        0,1,2,3,4,5,6,7,8,9,10,11,12,13,14,15);
      // split-f16 compensation: dot ~= Ah*Bh + Ah*Bl + Al*Bh (fp32-grade)
      acc = __builtin_amdgcn_wmma_f32_16x16x32_f16(false, ah, false, bh[c], (short)0, acc, false, false);
      acc = __builtin_amdgcn_wmma_f32_16x16x32_f16(false, ah, false, bl[c], (short)0, acc, false, false);
      acc = __builtin_amdgcn_wmma_f32_16x16x32_f16(false, al, false, bh[c], (short)0, acc, false, false);
    }

    // C/D layout: lane holds row N=nlo, codes M = 8*hi + r (r = VGPR index)
#pragma unroll
    for (int r = 0; r < 8; ++r) {
      int code = mt * 16 + 8 * hi + r;
      float d = lds_norm[code] - 2.0f * acc[r];
      if (d < best_d) { best_d = d; best_i = code; }   // ascending -> keeps first index
    }

#if USE_TDM
    if (wid == 0) __builtin_amdgcn_s_wait_tensorcnt(0);
#endif
    __syncthreads();
  }

  // merge lane <-> lane+16 (same x-row, disjoint code subsets)
  float od = __shfl_xor(best_d, 16, 32);
  int   oi = __shfl_xor(best_i, 16, 32);
  if (od < best_d || (od == best_d && oi < best_i)) { best_d = od; best_i = oi; }

  if (hi == 0) {
    int n = n0 + nlo;
    idx_ws[n] = best_i;
    idx_out[n] = (float)best_i;      // d_out is float-typed
  }
}

// ---------------------------------------------------------------------------
// Kernel 2: gather x_q -> transposed output + per-block loss partials
__global__ __launch_bounds__(256) void vq_gather(const float* __restrict__ x,
                                                 const float* __restrict__ emb,
                                                 const int* __restrict__ idx,
                                                 float* __restrict__ out,
                                                 float* __restrict__ partials) {
  __shared__ int   sidx[64];
  __shared__ float red[256];
  const int t  = threadIdx.x;
  const int b  = blockIdx.x >> 4;
  const int l0 = (blockIdx.x & 15) * 64;
  if (t < 64) sidx[t] = idx[b * Ln + l0 + t];
  __syncthreads();
  const int tl = t & 63, tc = t >> 6;
  const float* erow = emb + (size_t)sidx[tl] * Cn;
  const size_t base = (size_t)b * Cn * Ln + l0 + tl;
  float local = 0.f;
#pragma unroll 4
  for (int c = tc; c < Cn; c += 4) {
    float e  = erow[c];
    float xv = x[base + (size_t)c * Ln];
    out[base + (size_t)c * Ln] = e;      // straight-through: x + (x_q - x) == x_q
    float df = e - xv;
    local += df * df;
  }
  red[t] = local;
  __syncthreads();
  for (int s = 128; s > 0; s >>= 1) {
    if (t < s) red[t] += red[t + s];
    __syncthreads();
  }
  if (t == 0) partials[blockIdx.x] = red[0];   // deterministic (no float atomics)
}

// Kernel 3: finalize vq_loss = (1 + 0.25) * mean((x_q - x)^2)
__global__ __launch_bounds__(256) void vq_final(const float* __restrict__ partials,
                                                float* __restrict__ out_scalar) {
  __shared__ float red[256];
  float s = 0.f;
  for (int i = threadIdx.x; i < GATHER_BLOCKS; i += 256) s += partials[i];
  red[threadIdx.x] = s;
  __syncthreads();
  for (int k = 128; k > 0; k >>= 1) {
    if (threadIdx.x < k) red[threadIdx.x] += red[threadIdx.x + k];
    __syncthreads();
  }
  if (threadIdx.x == 0) *out_scalar = 1.25f * red[0] / (float)(Nn * Cn);
}

// ---------------------------------------------------------------------------
extern "C" void kernel_launch(void* const* d_in, const int* in_sizes, int n_in,
                              void* d_out, int out_size, void* d_ws, size_t ws_size,
                              hipStream_t stream) {
  const float* x   = (const float*)d_in[0];   // [64,256,1024]
  const float* emb = (const float*)d_in[1];   // [1024,256]
  float* out = (float*)d_out;
  float* ws  = (float*)d_ws;

  // ws layout: enorm[1024] f32 | partials[1024] f32 | idx[65536] i32 |
  //            emb_hi[262144] f16 | emb_lo[262144] f16   (~1.3 MB total)
  float*    enorm    = ws;
  float*    partials = ws + 1024;
  int*      idx_ws   = (int*)(ws + 2048);
  _Float16* eh       = (_Float16*)(ws + 2048 + Nn);
  _Float16* el       = eh + (size_t)En * Cn;

  float* out_xq   = out;                       // 16,777,216 f32
  float* out_idx  = out + (size_t)Nn * Cn;     // 65,536 f32
  float* out_loss = out_idx + Nn;              // 1 f32

  vq_norms <<<En / 256, 256, 0, stream>>>(emb, enorm);
  vq_split <<<(En * Cn / 4) / 256, 256, 0, stream>>>(emb, eh, el);
  vq_argmin<<<Nn / ROWS_PER_BLOCK, 256, 0, stream>>>(x, eh, el, enorm, idx_ws, out_idx);
  vq_gather<<<GATHER_BLOCKS, 256, 0, stream>>>(x, emb, idx_ws, out_xq, partials);
  vq_final <<<1, 256, 0, stream>>>(partials, out_loss);
}
